// StrategyQuantizer_64647847739782
// MI455X (gfx1250) — matrix operand order repaired and checked by reading, not compiled, not run on hardware
//
#include <hip/hip_runtime.h>

typedef __attribute__((ext_vector_type(16))) __bf16 v16bf;
typedef __attribute__((ext_vector_type(8)))  float  v8f;

#define NROWS           262144
#define ROWS_PER_BLOCK  256                       // 8 waves x 32 rows
#define NCHUNKS         (NROWS / ROWS_PER_BLOCK)  // 1024
#define NBLOCKS         512                       // persistent; 2 chunks per block

// LDS: codebook bf16 hi/lo (row k: 64 hi then 64 lo halves) + exact f32 ||w_k||^2
#define LDS_CB_BYTES   (1024 * 128 * 2)           // 256 KB
#define LDS_W2_BYTES   (1024 * 4)                 // 4 KB
#define LDS_TOTAL      (LDS_CB_BYTES + LDS_W2_BYTES)

#define WMMA_BF16(A, B, C) \
  __builtin_amdgcn_wmma_f32_16x16x32_bf16(false, (A), false, (B), (short)0, (C), false, false)

__global__ void __launch_bounds__(256, 1)
vq_argmin_gather_kernel(const float* __restrict__ x,
                        const float* __restrict__ w,
                        float* __restrict__ out)
{
  extern __shared__ char smem[];
  __bf16* cb  = (__bf16*)smem;
  float*  w2s = (float*)(smem + LDS_CB_BYTES);

  const int tid   = threadIdx.x;     // 0..255 (8 waves)
  const int wave  = tid >> 5;        // 0..7
  const int lane  = tid & 31;
  const int lhalf = lane >> 4;       // 0: lanes 0-15, 1: lanes 16-31
  const int l16   = lane & 15;

  // ---- One-time: stage codebook into LDS as split bf16 (hi + residual lo) ----
  #pragma unroll 4
  for (int j = 0; j < 256; ++j) {
    int idx = j * 256 + tid;                 // coalesced over 65536 elements
    float v = w[idx];
    __bf16 hi = (__bf16)v;
    __bf16 lo = (__bf16)(v - (float)hi);
    int k = idx >> 6, d = idx & 63;
    cb[k * 128 + d]      = hi;
    cb[k * 128 + 64 + d] = lo;
  }
  // ---- One-time: exact f32 ||w_k||^2 ----
  for (int r = 0; r < 4; ++r) {
    int k = r * 256 + tid;
    const float* wr = w + (size_t)k * 64;
    float s = 0.f;
    #pragma unroll
    for (int d = 0; d < 64; ++d) s = fmaf(wr[d], wr[d], s);
    w2s[k] = s;
  }
  __syncthreads();   // codebook resident for the rest of the kernel

  const int dsel = lhalf * 16;   // B layout: lanes 0-15 -> K 0..15, lanes 16-31 -> K 16..31

  // ---- Persistent chunk loop: waves fully independent from here on ----
  for (int chunk = blockIdx.x; chunk < NCHUNKS; chunk += NBLOCKS) {
    const int rowBase = chunk * ROWS_PER_BLOCK + wave * 32;   // this wave: 32 rows

    // A fragments, two 16-row tiles. contiguous k order (this code's, same as B):
    //   lanes 0-15  -> row, elems {K0..15}
    //   lanes 16-31 -> row, elems {K16..31}
    v16bf ah[2][2], al[2][2];   // [row tile][K half]
    #pragma unroll
    for (int t = 0; t < 2; ++t) {
      const float* xr = x + (size_t)(rowBase + t * 16 + l16) * 64;
      const int dbase = lhalf * 16;
      #pragma unroll
      for (int h = 0; h < 2; ++h) {
        float tmp[16];
        #pragma unroll
        for (int e = 0; e < 8; ++e) tmp[e]     = xr[h * 32 + dbase + e];
        #pragma unroll
        for (int e = 0; e < 8; ++e) tmp[8 + e] = xr[h * 32 + dbase + 8 + e];
        v16bf hi, lo;
        #pragma unroll
        for (int e = 0; e < 16; ++e) {
          __bf16 hh = (__bf16)tmp[e];
          hi[e] = hh;
          lo[e] = (__bf16)(tmp[e] - (float)hh);
        }
        ah[t][h] = hi;
        al[t][h] = lo;
      }
    }

    float minv[2][8];
    int   mini[2][8];
    #pragma unroll
    for (int t = 0; t < 2; ++t)
      #pragma unroll
      for (int i = 0; i < 8; ++i) { minv[t][i] = 3.4e38f; mini[t][i] = 0; }

    // ---- Sweep 64 code tiles; 12 WMMAs per tile, B-fragment-major order ----
    for (int ct = 0; ct < 64; ++ct) {
      const int c = ct * 16 + l16;            // this lane's code column
      const __bf16* row = cb + c * 128;

      // Issue all LDS reads up front (w2 first: LDS returns in order, so the
      // epilogue's dependence is satisfied once the B fragments arrive).
      const float w2c = w2s[c];
      v16bf bh0 = *(const v16bf*)(row + dsel);        // hi, K-half 0
      v16bf bh1 = *(const v16bf*)(row + 32 + dsel);   // hi, K-half 1
      v16bf bl0 = *(const v16bf*)(row + 64 + dsel);   // lo, K-half 0
      v16bf bl1 = *(const v16bf*)(row + 96 + dsel);   // lo, K-half 1

      v8f acc0 = {0.f, 0.f, 0.f, 0.f, 0.f, 0.f, 0.f, 0.f};
      v8f acc1 = {0.f, 0.f, 0.f, 0.f, 0.f, 0.f, 0.f, 0.f};

      // dot ~= xh*wh + xh*wl + xl*wh (bf16x3, f32 acc); consume each B frag fully
      acc0 = WMMA_BF16(ah[0][0], bh0, acc0);
      acc1 = WMMA_BF16(ah[1][0], bh0, acc1);
      acc0 = WMMA_BF16(al[0][0], bh0, acc0);
      acc1 = WMMA_BF16(al[1][0], bh0, acc1);

      acc0 = WMMA_BF16(ah[0][1], bh1, acc0);
      acc1 = WMMA_BF16(ah[1][1], bh1, acc1);
      acc0 = WMMA_BF16(al[0][1], bh1, acc0);
      acc1 = WMMA_BF16(al[1][1], bh1, acc1);

      acc0 = WMMA_BF16(ah[0][0], bl0, acc0);
      acc1 = WMMA_BF16(ah[1][0], bl0, acc1);

      acc0 = WMMA_BF16(ah[0][1], bl1, acc0);
      acc1 = WMMA_BF16(ah[1][1], bl1, acc1);

      #pragma unroll
      for (int i = 0; i < 8; ++i) {
        float s0 = fmaf(-2.f, acc0[i], w2c);  // ||x||^2 dropped: argmin-invariant
        if (s0 < minv[0][i]) { minv[0][i] = s0; mini[0][i] = c; }
        float s1 = fmaf(-2.f, acc1[i], w2c);
        if (s1 < minv[1][i]) { minv[1][i] = s1; mini[1][i] = c; }
      }
    }

    // ---- Argmin butterfly across the 16 lanes of each column group ----
    #pragma unroll
    for (int t = 0; t < 2; ++t)
      #pragma unroll
      for (int i = 0; i < 8; ++i) {
        float v = minv[t][i];
        int   ix = mini[t][i];
        #pragma unroll
        for (int m = 1; m < 16; m <<= 1) {
          float ov = __shfl_xor(v, m, 32);
          int   oi = __shfl_xor(ix, m, 32);
          if (ov < v || (ov == v && oi < ix)) { v = ov; ix = oi; }
        }
        mini[t][i] = ix;   // now uniform across each 16-lane half
      }

    // ---- Direct gather, no LDS/no barrier: half-wave h owns rows M = 8h..8h+7
    // of each tile. Each of its 16 lanes copies 4 floats -> 256B segments.
    #pragma unroll
    for (int t = 0; t < 2; ++t)
      #pragma unroll
      for (int i = 0; i < 8; ++i) {
        const int k   = mini[t][i];
        const int row = rowBase + t * 16 + lhalf * 8 + i;
        const float4 v4 = *(const float4*)(w + (size_t)k * 64 + l16 * 4);
        *(float4*)(out + (size_t)row * 64 + l16 * 4) = v4;
      }
  }
}

extern "C" void kernel_launch(void* const* d_in, const int* in_sizes, int n_in,
                              void* d_out, int out_size, void* d_ws, size_t ws_size,
                              hipStream_t stream) {
  (void)in_sizes; (void)n_in; (void)d_ws; (void)ws_size; (void)out_size;
  const float* x = (const float*)d_in[0];   // [262144, 64] f32
  const float* w = (const float*)d_in[1];   // [1024, 64]  f32
  float* out = (float*)d_out;               // [262144, 64] f32

  hipFuncSetAttribute((const void*)vq_argmin_gather_kernel,
                      hipFuncAttributeMaxDynamicSharedMemorySize, LDS_TOTAL);

  dim3 grid(NBLOCKS);
  dim3 block(256);           // 8 wave32 waves
  vq_argmin_gather_kernel<<<grid, block, LDS_TOTAL, stream>>>(x, w, out);
}
